// Text_Encoder_85916525789533
// MI455X (gfx1250) — matrix operand order repaired and checked by reading, not compiled, not run on hardware
//
#include <hip/hip_runtime.h>
#include <cstdint>
#include <cstddef>

// Problem constants (match reference)
#define BATCH   32
#define LSEQ    256
#define WDIM    1024
#define DIN     2048      // D_INNER
#define DSTATE  16
#define DTRANK  64
#define EMBED   768
#define NROWS   (BATCH*LSEQ)   // 8192

typedef __bf16 bf16;
typedef __attribute__((ext_vector_type(16))) __bf16 v16bf;
typedef __attribute__((ext_vector_type(8)))  __bf16 v8bf;
typedef __attribute__((ext_vector_type(8)))  float  v8f;

__device__ __forceinline__ float fast_sigmoid(float x) { return 1.f / (1.f + __expf(-x)); }

// CDNA5 async global->LDS copy (16B), tracked by ASYNCcnt (ISA 08, VGLOBAL op 98).
__device__ __forceinline__ void async_copy_b128(uint32_t lds_byte_off, const bf16* gptr) {
  asm volatile("global_load_async_to_lds_b128 %0, %1, off"
               :: "v"(lds_byte_off), "v"(gptr)
               : "memory");
}
__device__ __forceinline__ void wait_async0() {
  asm volatile("s_wait_asynccnt 0x0" ::: "memory");
}
// generic pointer -> LDS byte offset (ISA 10.2: low 32 bits of the address)
__device__ __forceinline__ uint32_t lds_off(const void* p) {
  return (uint32_t)(uintptr_t)p;
}

// ---------------------------------------------------------------------------
// eot[b] = argmax(text[b,:]) (first max), L == 256 == blockDim
// ---------------------------------------------------------------------------
__global__ __launch_bounds__(256) void eot_kernel(const int* __restrict__ text,
                                                  int* __restrict__ eot) {
  __shared__ int sv[256];
  __shared__ int si[256];
  const int b = blockIdx.x, t = threadIdx.x;
  sv[t] = text[b * LSEQ + t];
  si[t] = t;
  __syncthreads();
  for (int s = 128; s > 0; s >>= 1) {
    if (t < s) {
      const int vo = sv[t + s], io = si[t + s];
      if (vo > sv[t] || (vo == sv[t] && io < si[t])) { sv[t] = vo; si[t] = io; }
    }
    __syncthreads();
  }
  if (t == 0) eot[b] = si[0];
}

// ---------------------------------------------------------------------------
// xb[b,l,w] = bf16(tok_emb[text[b,l]][w] + pos_emb[l][w])
// ---------------------------------------------------------------------------
__global__ __launch_bounds__(256) void embed_kernel(const int* __restrict__ text,
                                                    const float* __restrict__ tok,
                                                    const float* __restrict__ pos,
                                                    bf16* __restrict__ xb) {
  const int tid = blockIdx.x * 256 + threadIdx.x;   // over NROWS*WDIM
  const int w  = tid & (WDIM - 1);
  const int bl = tid >> 10;
  const int l  = bl & (LSEQ - 1);
  const int tk = text[bl];
  xb[tid] = (bf16)(tok[(size_t)tk * WDIM + w] + pos[l * WDIM + w]);
}

// ---------------------------------------------------------------------------
// generic f32 -> bf16 cast
// ---------------------------------------------------------------------------
__global__ __launch_bounds__(256) void cast_bf16_kernel(const float* __restrict__ src,
                                                        bf16* __restrict__ dst, int n) {
  const int tid = blockIdx.x * 256 + threadIdx.x;
  if (tid < n) dst[tid] = (bf16)src[tid];
}

// strided cast: dtpart[row, 0:64] = bf16(x_dbl[row, 0:64])  (x_dbl row stride 96)
__global__ __launch_bounds__(256) void dtpart_cast_kernel(const float* __restrict__ xdbl,
                                                          bf16* __restrict__ dtp) {
  const int tid = blockIdx.x * 256 + threadIdx.x;   // NROWS*DTRANK
  const int r = tid / DTRANK, c = tid - r * DTRANK;
  dtp[tid] = (bf16)xdbl[(size_t)r * 96 + c];
}

// ---------------------------------------------------------------------------
// Tiled WMMA NT-GEMM:  C[M,N] = A[M,K] * Bt[N,K]^T , A/Bt bf16, f32 accumulate.
// 8 waves/WG arranged WROWS x WCOLS, per-wave tile (TM*16) x (TN*16), BK=32.
// Global->LDS staging uses CDNA5 async-to-LDS copies with double buffering:
// tile k+1 is in flight (ASYNCcnt) while tile k is consumed by WMMAs.
// EPI: 0 -> f32 store, 1 -> bf16 store, 2 -> softplus(acc + bias) f32 store
// ---------------------------------------------------------------------------
template <int BM, int BN, int WROWS, int WCOLS, int TM, int TN, int EPI>
__global__ __launch_bounds__(256) void wmma_gemm_nt(const bf16* __restrict__ A,
                                                    const bf16* __restrict__ Bt,
                                                    float* __restrict__ C,
                                                    bf16* __restrict__ Cb,
                                                    const float* __restrict__ bias,
                                                    int M, int N, int K) {
  constexpr int BK = 32;
  static_assert(WROWS * TM * 16 == BM, "tile M");
  static_assert(WCOLS * TN * 16 == BN, "tile N");
  __shared__ bf16 As[2][BM * BK];
  __shared__ bf16 Bs[2][BN * BK];

  const int tid  = threadIdx.x;
  const int lane = tid & 31;
  const int wave = tid >> 5;
  const int wm   = (wave / WCOLS) * (TM * 16);
  const int wn   = (wave % WCOLS) * (TN * 16);
  const int m0   = blockIdx.y * BM;
  const int n0   = blockIdx.x * BN;
  const int half = lane >> 4;      // 0 / 1
  const int lr   = lane & 15;

  constexpr int ACH = (BM * BK) / 8;   // 16B chunks in A tile
  constexpr int BCH = (BN * BK) / 8;   // 16B chunks in B tile

  // Issue all async copies for one (A,B) tile pair into buffer `buf`.
  auto issue_tile = [&](int buf, int k0) {
#pragma unroll
    for (int it = 0; it < (ACH + 255) / 256; ++it) {
      const int i = tid + it * 256;
      if ((ACH % 256 == 0) || (i < ACH)) {
        const int row = i >> 2, ks = (i & 3) * 8;
        async_copy_b128(lds_off(&As[buf][row * BK + ks]),
                        &A[(size_t)(m0 + row) * K + k0 + ks]);
      }
    }
#pragma unroll
    for (int it = 0; it < (BCH + 255) / 256; ++it) {
      const int i = tid + it * 256;
      if ((BCH % 256 == 0) || (i < BCH)) {
        const int row = i >> 2, ks = (i & 3) * 8;
        async_copy_b128(lds_off(&Bs[buf][row * BK + ks]),
                        &Bt[(size_t)(n0 + row) * K + k0 + ks]);
      }
    }
  };

  v8f acc[TM][TN];
#pragma unroll
  for (int i = 0; i < TM; ++i)
#pragma unroll
    for (int j = 0; j < TN; ++j)
#pragma unroll
      for (int e = 0; e < 8; ++e) acc[i][j][e] = 0.f;

  // prologue: stage tile 0
  issue_tile(0, 0);
  wait_async0();
  __syncthreads();

  const int NK = K / BK;
  for (int kt = 0; kt < NK; ++kt) {
    const int cur = kt & 1;
    if (kt + 1 < NK) issue_tile(cur ^ 1, (kt + 1) * BK);   // prefetch next tile

    v16bf af[TM];
    v16bf bfv[TN];
#pragma unroll
    for (int i = 0; i < TM; ++i) {   // A 16x32: lane<16 -> K 0..15 ; lane>=16 -> K 16..31
      const int row = wm + i * 16 + lr;
      const v8bf lo = *(const v8bf*)&As[cur][row * BK + half * 16];
      const v8bf hi = *(const v8bf*)&As[cur][row * BK + half * 16 + 8];
#pragma unroll
      for (int e = 0; e < 8; ++e) { af[i][e] = lo[e]; af[i][8 + e] = hi[e]; }
    }
#pragma unroll
    for (int j = 0; j < TN; ++j) {   // B 32x16: lane<16 -> K 0..15 ; lane>=16 -> K 16..31
      const int col = wn + j * 16 + lr;
      bfv[j] = *(const v16bf*)&Bs[cur][col * BK + half * 16];
    }
#pragma unroll
    for (int i = 0; i < TM; ++i)
#pragma unroll
      for (int j = 0; j < TN; ++j)
        acc[i][j] = __builtin_amdgcn_wmma_f32_16x16x32_bf16(
            false, af[i], false, bfv[j], (short)0, acc[i][j], false, false);

    wait_async0();      // my prefetch into the other buffer has landed
    __syncthreads();    // all waves done reading `cur` / writing `cur^1`
  }

  // Epilogue per ISA C/D layout: VGPR r -> M = r + (lane>=16 ? 8 : 0), N = lane&15
  const int rbase = half * 8;
#pragma unroll
  for (int i = 0; i < TM; ++i) {
#pragma unroll
    for (int j = 0; j < TN; ++j) {
      const int col = n0 + wn + j * 16 + lr;
#pragma unroll
      for (int r = 0; r < 8; ++r) {
        const int row = m0 + wm + i * 16 + rbase + r;
        const float v = acc[i][j][r];
        const size_t idx = (size_t)row * N + col;
        if (EPI == 0) {
          C[idx] = v;
        } else if (EPI == 1) {
          Cb[idx] = (bf16)v;
        } else {
          const float t = v + bias[col];
          C[idx] = (t > 20.f) ? t : log1pf(__expf(t));   // softplus
        }
      }
    }
  }
}

// ---------------------------------------------------------------------------
// depthwise causal conv(4) + bias + SiLU over xm = xz[:, :, 0:DIN]
// ---------------------------------------------------------------------------
__global__ __launch_bounds__(256) void conv_silu_kernel(const bf16* __restrict__ xz,
                                                        const float* __restrict__ cw,
                                                        const float* __restrict__ cb,
                                                        bf16* __restrict__ xcb) {
  const int tid = blockIdx.x * 256 + threadIdx.x;  // NROWS*DIN
  const int d  = tid & (DIN - 1);
  const int bl = tid / DIN;
  const int l  = bl & (LSEQ - 1);
  float acc = cb[d];
#pragma unroll
  for (int j = 0; j < 4; ++j) {
    const int ls = l - 3 + j;
    if (ls >= 0)
      acc += cw[d * 4 + j] * (float)xz[(size_t)(bl - 3 + j) * (2 * DIN) + d];
  }
  xcb[(size_t)bl * DIN + d] = (bf16)(acc * fast_sigmoid(acc));
}

// ---------------------------------------------------------------------------
// selective scan: per (b,d) recurrence over L; emit gated y only at l==eot[b]
// ---------------------------------------------------------------------------
__global__ __launch_bounds__(256) void scan_kernel(const float* __restrict__ dt,
                                                   const float* __restrict__ xdbl,
                                                   const bf16* __restrict__ xcb,
                                                   const bf16* __restrict__ xz,
                                                   const float* __restrict__ alog,
                                                   const float* __restrict__ dpar,
                                                   const int* __restrict__ eot,
                                                   float* __restrict__ pooled_y) {
  const int b = blockIdx.x;
  const int d = blockIdx.y * 256 + threadIdx.x;
  float A[DSTATE], h[DSTATE];
#pragma unroll
  for (int s = 0; s < DSTATE; ++s) {
    A[s] = -__expf(alog[d * DSTATE + s]);
    h[s] = 0.f;
  }
  const float dp = dpar[d];
  const int eb = eot[b];
  __shared__ float sBC[2 * DSTATE];

  for (int l = 0; l < LSEQ; ++l) {
    __syncthreads();
    if (threadIdx.x < 2 * DSTATE)
      sBC[threadIdx.x] = xdbl[(size_t)(b * LSEQ + l) * 96 + DTRANK + threadIdx.x];
    __syncthreads();
    const size_t base = (size_t)(b * LSEQ + l) * DIN + d;
    const float dtv = dt[base];
    const float u   = (float)xcb[base];
    float y = 0.f;
#pragma unroll
    for (int s = 0; s < DSTATE; ++s) {
      const float dA = __expf(dtv * A[s]);
      h[s] = dA * h[s] + dtv * sBC[s] * u;
      y += h[s] * sBC[DSTATE + s];
    }
    if (l == eb) {
      const float z  = (float)xz[(size_t)(b * LSEQ + l) * (2 * DIN) + DIN + d];
      pooled_y[b * DIN + d] = (y + u * dp) * (z * fast_sigmoid(z));
    }
  }
}

// ---------------------------------------------------------------------------
// px[b,n] = sum_k pooled_y[b,k] * out_proj_w[n,k]   (32x1024, K=2048)
// ---------------------------------------------------------------------------
__global__ __launch_bounds__(256) void pooled_outproj_kernel(const float* __restrict__ py,
                                                             const float* __restrict__ ow,
                                                             float* __restrict__ px) {
  const int tid = blockIdx.x * 256 + threadIdx.x;  // BATCH*WDIM
  const int b = tid >> 10, n = tid & (WDIM - 1);
  const float4* a = (const float4*)(py + (size_t)b * DIN);
  const float4* w = (const float4*)(ow + (size_t)n * DIN);
  float acc = 0.f;
  for (int k = 0; k < DIN / 4; ++k) {
    const float4 av = a[k], wv = w[k];
    acc += av.x * wv.x + av.y * wv.y + av.z * wv.z + av.w * wv.w;
  }
  px[tid] = acc;
}

// ---------------------------------------------------------------------------
// LayerNorm(32x1024) then out[b,e] = sum_w xn[b,w] * text_proj[w,e]
// ---------------------------------------------------------------------------
__global__ __launch_bounds__(256) void ln_final_kernel(const float* __restrict__ px,
                                                       const float* __restrict__ lnw,
                                                       const float* __restrict__ lnb,
                                                       const float* __restrict__ tp,
                                                       float* __restrict__ out) {
  __shared__ float red[256];
  __shared__ float xn[WDIM];
  const int b = blockIdx.x, t = threadIdx.x;
  const float* xr = px + (size_t)b * WDIM;

  float s = 0.f;
  for (int w = t; w < WDIM; w += 256) s += xr[w];
  red[t] = s; __syncthreads();
  for (int st = 128; st > 0; st >>= 1) { if (t < st) red[t] += red[t + st]; __syncthreads(); }
  const float mu = red[0] * (1.f / WDIM);
  __syncthreads();

  float v = 0.f;
  for (int w = t; w < WDIM; w += 256) { const float dv = xr[w] - mu; v += dv * dv; }
  red[t] = v; __syncthreads();
  for (int st = 128; st > 0; st >>= 1) { if (t < st) red[t] += red[t + st]; __syncthreads(); }
  const float rstd = rsqrtf(red[0] * (1.f / WDIM) + 1e-5f);
  __syncthreads();

  for (int w = t; w < WDIM; w += 256) xn[w] = (xr[w] - mu) * rstd * lnw[w] + lnb[w];
  __syncthreads();

  for (int e = t; e < EMBED; e += 256) {
    float acc = 0.f;
    for (int w = 0; w < WDIM; ++w) acc += xn[w] * tp[(size_t)w * EMBED + e];
    out[b * EMBED + e] = acc;
  }
}

// ---------------------------------------------------------------------------
extern "C" void kernel_launch(void* const* d_in, const int* in_sizes, int n_in,
                              void* d_out, int out_size, void* d_ws, size_t ws_size,
                              hipStream_t stream) {
  const int*   text   = (const int*)d_in[0];
  const float* tok    = (const float*)d_in[1];
  const float* pos    = (const float*)d_in[2];
  const float* inpw   = (const float*)d_in[3];
  const float* convw  = (const float*)d_in[4];
  const float* convb  = (const float*)d_in[5];
  const float* xprojw = (const float*)d_in[6];
  const float* dtpw   = (const float*)d_in[7];
  const float* dtpb   = (const float*)d_in[8];
  const float* alog   = (const float*)d_in[9];
  const float* dpar   = (const float*)d_in[10];
  const float* outw   = (const float*)d_in[11];
  const float* lnw    = (const float*)d_in[12];
  const float* lnb    = (const float*)d_in[13];
  const float* tproj  = (const float*)d_in[14];
  float* out = (float*)d_out;

  char* p = (char*)d_ws;
  auto alloc = [&](size_t bytes) -> char* {
    char* r = p;
    p += (bytes + 255) & ~(size_t)255;
    return r;
  };
  bf16* xb     = (bf16*)alloc((size_t)NROWS * WDIM * sizeof(bf16));      // 16.8 MB
  bf16* w1b    = (bf16*)alloc((size_t)(2 * DIN) * WDIM * sizeof(bf16));  //  8.4 MB
  bf16* w2b    = (bf16*)alloc((size_t)96 * DIN * sizeof(bf16));          //  0.4 MB
  bf16* w3b    = (bf16*)alloc((size_t)DIN * DTRANK * sizeof(bf16));      //  0.3 MB
  bf16* xz     = (bf16*)alloc((size_t)NROWS * (2 * DIN) * sizeof(bf16)); // 67.1 MB
  bf16* xcb    = (bf16*)alloc((size_t)NROWS * DIN * sizeof(bf16));       // 33.6 MB
  float* xdbl  = (float*)alloc((size_t)NROWS * 96 * sizeof(float));      //  3.1 MB
  bf16* dtp    = (bf16*)alloc((size_t)NROWS * DTRANK * sizeof(bf16));    //  1.0 MB
  float* dtf   = (float*)alloc((size_t)NROWS * DIN * sizeof(float));     // 67.1 MB
  float* py    = (float*)alloc((size_t)BATCH * DIN * sizeof(float));
  float* px    = (float*)alloc((size_t)BATCH * WDIM * sizeof(float));
  int*   eot   = (int*)alloc(BATCH * sizeof(int));

  // 1) pooling indices
  eot_kernel<<<BATCH, 256, 0, stream>>>(text, eot);

  // 2) embedding -> bf16 activations
  embed_kernel<<<(NROWS * WDIM) / 256, 256, 0, stream>>>(text, tok, pos, xb);

  // 3) weight casts to bf16
  {
    int n1 = 2 * DIN * WDIM, n2 = 96 * DIN, n3 = DIN * DTRANK;
    cast_bf16_kernel<<<(n1 + 255) / 256, 256, 0, stream>>>(inpw, w1b, n1);
    cast_bf16_kernel<<<(n2 + 255) / 256, 256, 0, stream>>>(xprojw, w2b, n2);
    cast_bf16_kernel<<<(n3 + 255) / 256, 256, 0, stream>>>(dtpw, w3b, n3);
  }

  // 4) in_proj: xz(bf16) = xb(8192x1024) @ in_proj_w^T (4096x1024)
  wmma_gemm_nt<128, 128, 2, 4, 4, 2, 1>
      <<<dim3((2 * DIN) / 128, NROWS / 128), 256, 0, stream>>>(
          xb, w1b, nullptr, xz, nullptr, NROWS, 2 * DIN, WDIM);

  // 5) depthwise conv + SiLU -> xcb(bf16)
  conv_silu_kernel<<<(NROWS * DIN) / 256, 256, 0, stream>>>(xz, convw, convb, xcb);

  // 6) x_proj: x_dbl(f32 8192x96) = xcb @ x_proj_w^T
  wmma_gemm_nt<128, 96, 4, 2, 2, 3, 0>
      <<<dim3(1, NROWS / 128), 256, 0, stream>>>(
          xcb, w2b, xdbl, nullptr, nullptr, NROWS, 96, DIN);

  // 7) dt rank slice -> bf16
  dtpart_cast_kernel<<<(NROWS * DTRANK) / 256, 256, 0, stream>>>(xdbl, dtp);

  // 8) dt = softplus(dtp @ dt_proj_w^T + dt_proj_b)  (8192x2048, K=64)
  wmma_gemm_nt<128, 128, 2, 4, 4, 2, 2>
      <<<dim3(DIN / 128, NROWS / 128), 256, 0, stream>>>(
          dtp, w3b, dtf, nullptr, dtpb, NROWS, DIN, DTRANK);

  // 9) selective scan; emits gated pooled rows only at l == eot[b]
  scan_kernel<<<dim3(BATCH, DIN / 256), 256, 0, stream>>>(
      dtf, xdbl, xcb, xz, alog, dpar, eot, py);

  // 10) out_proj on the 32 pooled rows only (f32 for final precision)
  pooled_outproj_kernel<<<(BATCH * WDIM) / 256, 256, 0, stream>>>(py, outw, px);

  // 11) LayerNorm + final text projection
  ln_final_kernel<<<BATCH, 256, 0, stream>>>(px, lnw, lnb, tproj, out);
}
